// E3NNLiftingTensorProduct_21251498181473
// MI455X (gfx1250) — matrix operand order repaired and checked by reading, not compiled, not run on hardware
//
#include <hip/hip_runtime.h>

typedef __attribute__((ext_vector_type(16))) _Float16 v16h;
typedef __attribute__((ext_vector_type(8)))  _Float16 v8h;
typedef __attribute__((ext_vector_type(8)))  float    v8f;
typedef __attribute__((ext_vector_type(4)))  float    v4f;
typedef int v4i __attribute__((vector_size(4 * sizeof(int))));

#define DIN      576
#define DOUT     2304
#define NTOT     100000
#define BN       32                  // batch rows per block
#define PAD      72                  // halfs per LDS row -> 144B stride, bank-conflict-free
#define XS_HALFS (288 * PAD)         // (32*1 + 32*3 + 32*5) rows
#define RAW_FLT  (BN * DIN)          // 18432 floats = 73728 B raw staging
#define NBLOCKS  (NTOT / BN)         // 3125
#define GRID     625                 // 5 blocks per persistent workgroup (exact)

#if __has_builtin(__builtin_amdgcn_global_load_async_to_lds_b128)
#define USE_ASYNC 1
#else
#define USE_ASYNC 0
#endif

#if USE_ASYNC
__device__ __forceinline__ void async_copy16(const float* gsrc, float* lds)
{
    __builtin_amdgcn_global_load_async_to_lds_b128(
        (__attribute__((address_space(1))) v4i*)(__attribute__((address_space(1))) void*)gsrc,
        (__attribute__((address_space(3))) v4i*)(__attribute__((address_space(3))) void*)lds,
        0, 0);
}
#endif

__device__ __forceinline__ void async_wait0()
{
#if __has_builtin(__builtin_amdgcn_s_wait_asynccnt)
    __builtin_amdgcn_s_wait_asynccnt(0);
#else
    asm volatile("s_wait_asynccnt 0" ::: "memory");
#endif
    asm volatile("" ::: "memory");   // keep the LDS reads below the wait
}

// convert 4 staged fp32 x-values to f16 and scatter into the transposed
// [(n_loc,i)][u] LDS layout (sections: l=0 rows [0,32), l=1 [32,128), l=2 [128,288))
__device__ __forceinline__ void scatter_vals(_Float16* __restrict__ xs, int k, int t, v4f val)
{
    const unsigned idx = (unsigned)(k * 256 + t) * 4u;
    const unsigned nl  = idx / DIN;
    const unsigned j   = idx - nl * DIN;
#pragma unroll
    for (unsigned e = 0; e < 4; ++e) {
        const unsigned jj = j + e;            // stays in-row: 576 % 4 == 0
        unsigned row, u;
        if (jj < 64u) {                       // l=0, d=1
            u   = jj;
            row = nl;
        } else if (jj < 256u) {               // l=1, d=3
            const unsigned q = jj - 64u;
            u   = q / 3u;
            row = 32u + nl * 3u + (q - u * 3u);
        } else {                              // l=2, d=5
            const unsigned q = jj - 256u;
            u   = q / 5u;
            row = 128u + nl * 5u + (q - u * 5u);
        }
        xs[row * PAD + u] = (_Float16)val[e];
    }
}

template <int D>
__device__ __forceinline__ void compute_l(const _Float16* __restrict__ xs,  // l section
                                          const v16h (&B)[2][2],            // reg-resident W slice
                                          float* __restrict__ out,          // d_out + ooff_l
                                          unsigned n0, int ln, int h, int w0)
{
    const int MT = 2 * D;  // 16-row M tiles in this l section (BN=32)
    for (int m = 0; m < MT; ++m) {
        // A fragment: lane half h holds K = 8h+0..7 (elems 0-7), 16+8h+0..7 (elems 8-15)
        const _Float16* xrow = xs + (m * 16 + ln) * PAD;
        v16h a0, a1;
        *((v8h*)&a0)     = *(const v8h*)(xrow + h * 8);
        *((v8h*)&a0 + 1) = *(const v8h*)(xrow + 16 + h * 8);
        *((v8h*)&a1)     = *(const v8h*)(xrow + 32 + h * 8);
        *((v8h*)&a1 + 1) = *(const v8h*)(xrow + 48 + h * 8);

#pragma unroll
        for (int tt = 0; tt < 2; ++tt) {
            v8f acc = {};
            acc = __builtin_amdgcn_wmma_f32_16x16x32_f16(false, a0, false, B[0][tt],
                                                         (short)0, acc, false, false);
            acc = __builtin_amdgcn_wmma_f32_16x16x32_f16(false, a1, false, B[1][tt],
                                                         (short)0, acc, false, false);
            const unsigned w = (unsigned)(w0 + 16 * tt + ln);  // C/D: N = lane&15
#pragma unroll
            for (int v = 0; v < 8; ++v) {
                const unsigned row = (unsigned)(m * 16 + 8 * h + v);  // C/D: M = v + 8h
                const unsigned nl  = row / (unsigned)D;
                const unsigned i   = row - nl * (unsigned)D;
                // 32-bit index: max 2.3e8 elements / 921MB byte offset -> fits u32,
                // enables saddr+voffset store form (no 64-bit per-lane address math)
                out[(n0 + nl) * (unsigned)DOUT + w * (unsigned)D + i] = acc[v] * 0.125f;
            }
        }
    }
}

__global__ __launch_bounds__(256) void e3nn_tp_kernel(const float* __restrict__ x,
                                                      const float* __restrict__ W0,
                                                      const float* __restrict__ W1,
                                                      const float* __restrict__ W2,
                                                      float* __restrict__ out)
{
#if USE_ASYNC
    __shared__ __align__(16) float raw[RAW_FLT];          // 73728 B async landing zone
#endif
    __shared__ __align__(16) _Float16 xs2[2][XS_HALFS];   // 2 x 41472 B f16 tiles

    const int t    = threadIdx.x;
    const int lane = t & 31;
    const int h    = lane >> 4;
    const int ln   = lane & 15;
    const int w0   = (t >> 5) * 32;   // 8 waves -> disjoint 32-wide w slices

    // ---- one-time: this wave's B fragments (W slice) straight into VGPRs ----
    // B layout: lane half h, element e -> K = 8h + e (e<8), 16 + 8h + (e-8) (e>=8) (+32 per k-step), N = lane&15
    v16h Bf[3][2][2];
    const float* Wl[3] = {W0, W1, W2};
#pragma unroll
    for (int l = 0; l < 3; ++l)
#pragma unroll
        for (int s = 0; s < 2; ++s)
#pragma unroll
            for (int tt = 0; tt < 2; ++tt) {
                const float* src = Wl[l] + (32 * s + 8 * h) * 256 + (w0 + 16 * tt + ln);
                v16h b;
#pragma unroll
                for (int e = 0; e < 16; ++e)
                    b[e] = (_Float16)src[(e < 8 ? e : e + 8) * 256];   // 64B-coalesced across lanes
                Bf[l][s][tt] = b;
            }

    // ---- prologue: stage first block into buffer 0 ----
    {
        const float* src = x + (unsigned)blockIdx.x * (BN * DIN);  // tile contiguous in x
#if USE_ASYNC
        for (int k = 0; k < 18; ++k) {
            const unsigned idx = (unsigned)(k * 256 + t) * 4u;
            async_copy16(src + idx, raw + idx);
        }
        async_wait0();
        for (int k = 0; k < 18; ++k)
            scatter_vals(xs2[0], k, t, *(const v4f*)(raw + (k * 256 + t) * 4));
#else
        for (int k = 0; k < 18; ++k)
            scatter_vals(xs2[0], k, t, *(const v4f*)(src + (unsigned)(k * 256 + t) * 4u));
#endif
    }
    __syncthreads();

    // ---- persistent, double-buffered block loop ----
    int cur = 0;
    for (int b = blockIdx.x; b < NBLOCKS; b += GRID) {
        const int  nb        = b + GRID;
        const bool have_next = nb < NBLOCKS;

#if USE_ASYNC
        if (have_next) {   // stream next raw tile into LDS during compute (ASYNCcnt)
            const float* src = x + (unsigned)nb * (BN * DIN);
            for (int k = 0; k < 18; ++k) {
                const unsigned idx = (unsigned)(k * 256 + t) * 4u;
                async_copy16(src + idx, raw + idx);
            }
        }
#else
        v4f pf[18];
        if (have_next) {   // VGPR prefetch fallback
            const float* src = x + (unsigned)nb * (BN * DIN);
#pragma unroll
            for (int k = 0; k < 18; ++k)
                pf[k] = *(const v4f*)(src + (unsigned)(k * 256 + t) * 4u);
        }
#endif

        const _Float16* xs = xs2[cur];
        const unsigned  n0 = (unsigned)b * BN;
        compute_l<1>(xs,             Bf[0], out,        n0, ln, h, w0);
        compute_l<3>(xs + 32 * PAD,  Bf[1], out + 256,  n0, ln, h, w0);
        compute_l<5>(xs + 128 * PAD, Bf[2], out + 1024, n0, ln, h, w0);

        if (have_next) {   // convert own bytes into the other f16 buffer
            _Float16* xn = xs2[cur ^ 1];
#if USE_ASYNC
            async_wait0();
            for (int k = 0; k < 18; ++k)
                scatter_vals(xn, k, t, *(const v4f*)(raw + (k * 256 + t) * 4));
#else
#pragma unroll
            for (int k = 0; k < 18; ++k)
                scatter_vals(xn, k, t, pf[k]);
#endif
        }
        __syncthreads();
        cur ^= 1;
    }
}

extern "C" void kernel_launch(void* const* d_in, const int* in_sizes, int n_in,
                              void* d_out, int out_size, void* d_ws, size_t ws_size,
                              hipStream_t stream)
{
    const float* x   = (const float*)d_in[0];
    const float* W0  = (const float*)d_in[1];
    const float* W1  = (const float*)d_in[2];
    const float* W2  = (const float*)d_in[3];
    float*       out = (float*)d_out;

    e3nn_tp_kernel<<<GRID, 256, 0, stream>>>(x, W0, W1, W2, out);
}
